// Network_72395968741514
// MI455X (gfx1250) — matrix operand order repaired and checked by reading, not compiled, not run on hardware
//
#include <hip/hip_runtime.h>
#include <math.h>

typedef __attribute__((ext_vector_type(16))) __bf16 v16bf;
typedef __attribute__((ext_vector_type(8)))  __bf16 v8bf;
typedef __attribute__((ext_vector_type(8)))  float  v8f;

#define NN        2048
#define BB        512
#define NUM_IN    256
#define NUM_OUT   128
#define MAX_ITERS 200
#define CONVT     0.1f
#define DAMP      0.8f
#define DIVT      10000.0f

#define BM      128
#define BN      64
#define BK      64
#define THREADS 256
#define LDSROW  72          // bf16 elements per LDS row (64 data + 8 pad = 144B stride)

struct Flags { unsigned done; unsigned parity; unsigned maxdiff; unsigned maxabs; };

// ---- W f32 -> bf16 (once per call) ------------------------------------------
__global__ void conv_w(const float* __restrict__ W, __bf16* __restrict__ Wb) {
    int i = (blockIdx.x * blockDim.x + threadIdx.x) * 4;
    if (i < NN * NN) {
        float4 v = *(const float4*)(W + i);
        Wb[i + 0] = (__bf16)v.x;
        Wb[i + 1] = (__bf16)v.y;
        Wb[i + 2] = (__bf16)v.z;
        Wb[i + 3] = (__bf16)v.w;
    }
}

// ---- state init: out0 = 0 with input columns clamped to x -------------------
__global__ void init_state(const float* __restrict__ x, float* __restrict__ buf0,
                           Flags* __restrict__ f) {
    int i = blockIdx.x * blockDim.x + threadIdx.x;
    if (i < BB * NN) {
        int n = i & (NN - 1);
        buf0[i] = (n < NUM_IN) ? x[i] : 0.0f;
    }
    if (i == 0) { f->done = 0; f->parity = 0; f->maxdiff = 0; f->maxabs = 0; }
}

// ---- one fixed-point iteration: GEMM (bf16 WMMA) + activations + diff max ---
__global__ __launch_bounds__(THREADS) void iter_kernel(
    const float*  __restrict__ cur, float* __restrict__ nxt,
    const __bf16* __restrict__ Wb,  const float* __restrict__ bias,
    const float*  __restrict__ x,   Flags* __restrict__ f) {

    if (*(volatile unsigned*)&f->done) return;   // uniform early-out once converged

    const int tid  = threadIdx.x;
    const int lane = tid & 31;
    const int wave = tid >> 5;
    const int n0   = blockIdx.x * BN;

    float dmax = 0.0f, amax = 0.0f;

    // ---------- input-neuron column blocks: W rows are zero, result is the
    // external clamp. No GEMM needed (uniform per block: n0 is scalar). ------
    if (n0 + BN <= NUM_IN) {
#pragma unroll
        for (int r = 0; r < (BM * BN) / THREADS; ++r) {
            int    idx = tid + r * THREADS;
            int    m   = blockIdx.y * BM + idx / BN;
            int    n   = n0 + (idx % BN);
            size_t g   = (size_t)m * NN + n;
            float  nv  = x[g];
            float  old = cur[g];
            dmax = fmaxf(dmax, fabsf(nv - old));
            amax = fmaxf(amax, fabsf(nv));
            nxt[g] = nv;
        }
#pragma unroll
        for (int s = 16; s > 0; s >>= 1) {
            dmax = fmaxf(dmax, __shfl_xor(dmax, s, 32));
            amax = fmaxf(amax, __shfl_xor(amax, s, 32));
        }
        if (lane == 0) {
            atomicMax(&f->maxdiff, __float_as_uint(dmax));
            atomicMax(&f->maxabs,  __float_as_uint(amax));
        }
        return;
    }

    // ---------- main WMMA path ----------
    __shared__ __align__(128) __bf16 sB[2][BN * LDSROW];   // 144B row stride: bank-conflict-free

    const int m0 = blockIdx.y * BM + wave * 16;

    v8f acc0 = {}, acc1 = {}, acc2 = {}, acc3 = {};

    const unsigned long long wbase = (unsigned long long)(uintptr_t)Wb;
    unsigned ldsBase[2];
    ldsBase[0] = (unsigned)(uintptr_t)(void*)&sB[0][0];
    ldsBase[1] = (unsigned)(uintptr_t)(void*)&sB[1][0];

    // async fill of one 64x64 bf16 W-tile: 512 x 16B chunks, 2 per thread
    auto issue_tile = [&](int t, int buf) {
#pragma unroll
        for (int c = 0; c < 2; ++c) {
            int chunk = tid + c * THREADS;
            int row   = chunk >> 3;          // 0..63 (local neuron row)
            int cc    = chunk & 7;           // 16B chunk within the 128B of data
            unsigned goff  = (unsigned)(((n0 + row) * NN + t * BK) * 2 + cc * 16);
            unsigned laddr = ldsBase[buf] + (unsigned)(row * (LDSROW * 2) + cc * 16);
            asm volatile("global_load_async_to_lds_b128 %0, %1, %2"
                         :: "v"(laddr), "v"(goff), "s"(wbase) : "memory");
        }
    };

    const int T = NN / BK;                   // 32 K-tiles
    issue_tile(0, 0);

    for (int t = 0; t < T; ++t) {
        const int buf = t & 1;
        if (t + 1 < T) {
            issue_tile(t + 1, buf ^ 1);
            asm volatile("s_wait_asynccnt 0x2" ::: "memory");   // tile t landed
        } else {
            asm volatile("s_wait_asynccnt 0x0" ::: "memory");
        }
        __syncthreads();

        const v8bf* p8 = (const v8bf*)&sB[buf][0];

#pragma unroll
        for (int kk = 0; kk < BK; kk += 32) {
            // ---- A fragment: row m, f32 -> bf16 (16-bit A 16x32 layout) ----
            int mrow = m0 + (lane & 15);
            int kb   = t * BK + kk + ((lane >> 4) << 4);
            const float4* pa = (const float4*)(cur + (size_t)mrow * NN + kb);
            float4 a0 = pa[0], a1 = pa[1];   // K = kb .. kb+7
            float4 a2 = pa[2], a3 = pa[3];   // K = kb+8 .. kb+15
            v16bf af;
            af[0]  = (__bf16)a0.x; af[1]  = (__bf16)a0.y;
            af[2]  = (__bf16)a0.z; af[3]  = (__bf16)a0.w;
            af[4]  = (__bf16)a1.x; af[5]  = (__bf16)a1.y;
            af[6]  = (__bf16)a1.z; af[7]  = (__bf16)a1.w;
            af[8]  = (__bf16)a2.x; af[9]  = (__bf16)a2.y;
            af[10] = (__bf16)a2.z; af[11] = (__bf16)a2.w;
            af[12] = (__bf16)a3.x; af[13] = (__bf16)a3.y;
            af[14] = (__bf16)a3.z; af[15] = (__bf16)a3.w;

            // ---- B fragments from LDS (lane: col n, K = 16*(lane>>4)+kk..) --
            int nl = lane & 15;
            int kO = kk + ((lane >> 4) << 4);           // 0/16/32/48
            // v8bf index: ((n_local)*LDSROW + kO) / 8  = n_local*9 + kO/8
            int i0 = (nl +  0) * (LDSROW / 8) + (kO >> 3);
            int i1 = (nl + 16) * (LDSROW / 8) + (kO >> 3);
            int i2 = (nl + 32) * (LDSROW / 8) + (kO >> 3);
            int i3 = (nl + 48) * (LDSROW / 8) + (kO >> 3);
            v16bf b0 = __builtin_shufflevector(p8[i0], p8[i0 + 1],
                        0,1,2,3,4,5,6,7,8,9,10,11,12,13,14,15);
            v16bf b1 = __builtin_shufflevector(p8[i1], p8[i1 + 1],
                        0,1,2,3,4,5,6,7,8,9,10,11,12,13,14,15);
            v16bf b2 = __builtin_shufflevector(p8[i2], p8[i2 + 1],
                        0,1,2,3,4,5,6,7,8,9,10,11,12,13,14,15);
            v16bf b3 = __builtin_shufflevector(p8[i3], p8[i3 + 1],
                        0,1,2,3,4,5,6,7,8,9,10,11,12,13,14,15);

            acc0 = __builtin_amdgcn_wmma_f32_16x16x32_bf16(false, af, false, b0,
                                                           (short)0, acc0, false, false);
            acc1 = __builtin_amdgcn_wmma_f32_16x16x32_bf16(false, af, false, b1,
                                                           (short)0, acc1, false, false);
            acc2 = __builtin_amdgcn_wmma_f32_16x16x32_bf16(false, af, false, b2,
                                                           (short)0, acc2, false, false);
            acc3 = __builtin_amdgcn_wmma_f32_16x16x32_bf16(false, af, false, b3,
                                                           (short)0, acc3, false, false);
        }
        __syncthreads();   // all waves done reading buf before it is refilled
    }

    // ---- epilogue: bias + activations + clamp + diff tracking ----
    const int nl = lane & 15;
    const int mb = (lane >> 4) << 3;
#pragma unroll
    for (int j = 0; j < 4; ++j) {
        v8f a = (j == 0) ? acc0 : (j == 1) ? acc1 : (j == 2) ? acc2 : acc3;
        int   n  = n0 + j * 16 + nl;
        float bn = bias[n];
#pragma unroll
        for (int r = 0; r < 8; ++r) {
            int   m  = m0 + mb + r;
            float ai = a[r] + bn;
            float nv;
            if (n >= NN - NUM_OUT)      nv = 1.0f / (1.0f + __expf(-ai));     // sigmoid
            else                        nv = DAMP * (ai > 0.0f ? ai : 0.01f * ai);
            if (n < NUM_IN)             nv = x[(size_t)m * NN + n];           // clamp
            float old = cur[(size_t)m * NN + n];
            dmax = fmaxf(dmax, fabsf(nv - old));
            amax = fmaxf(amax, fabsf(nv));
            nxt[(size_t)m * NN + n] = nv;
        }
    }
#pragma unroll
    for (int s = 16; s > 0; s >>= 1) {
        dmax = fmaxf(dmax, __shfl_xor(dmax, s, 32));
        amax = fmaxf(amax, __shfl_xor(amax, s, 32));
    }
    if (lane == 0) {                       // bits of non-negative floats order-compatible
        atomicMax(&f->maxdiff, __float_as_uint(dmax));
        atomicMax(&f->maxabs,  __float_as_uint(amax));
    }
}

// ---- convergence check (reference keep-condition) ---------------------------
__global__ void check_kernel(Flags* __restrict__ f, int iter) {
    if (f->done) return;
    float md = __uint_as_float(f->maxdiff);
    float ma = __uint_as_float(f->maxabs);
    bool keep = ((iter + 1) < MAX_ITERS) && (md >= CONVT) && (ma <= DIVT);
    f->parity = (unsigned)((iter + 1) & 1);     // which buffer holds the result
    if (!keep) f->done = 1;
    f->maxdiff = 0; f->maxabs = 0;
}

// ---- gather output columns --------------------------------------------------
__global__ void extract_kernel(const float* __restrict__ buf0,
                               const float* __restrict__ buf1,
                               const Flags* __restrict__ f, float* __restrict__ out) {
    int i = blockIdx.x * blockDim.x + threadIdx.x;
    if (i >= BB * NUM_OUT) return;
    int m = i / NUM_OUT, c = i % NUM_OUT;
    const float* src = f->parity ? buf1 : buf0;
    out[i] = src[(size_t)m * NN + (NN - NUM_OUT) + c];
}

extern "C" void kernel_launch(void* const* d_in, const int* in_sizes, int n_in,
                              void* d_out, int out_size, void* d_ws, size_t ws_size,
                              hipStream_t stream) {
    const float* x = (const float*)d_in[0];   // [B, N]
    const float* W = (const float*)d_in[1];   // [N, N]
    const float* b = (const float*)d_in[2];   // [N]
    // d_in[3]/d_in[4] = input_idx / output_idx (contiguous ranges, hardcoded)

    char*   ws   = (char*)d_ws;
    Flags*  f    = (Flags*)ws;
    __bf16* Wb   = (__bf16*)(ws + 1024);
    float*  buf0 = (float*)(ws + 1024 + (size_t)NN * NN * 2);
    float*  buf1 = buf0 + (size_t)BB * NN;
    float*  out  = (float*)d_out;

    conv_w<<<(NN * NN / 4 + 255) / 256, 256, 0, stream>>>(W, Wb);
    init_state<<<(BB * NN + 255) / 256, 256, 0, stream>>>(x, buf0, f);

    dim3 grid(NN / BN, BB / BM);              // 32 x 4
    for (int i = 0; i < MAX_ITERS; ++i) {
        float* curb = (i & 1) ? buf1 : buf0;
        float* nxtb = (i & 1) ? buf0 : buf1;
        iter_kernel<<<grid, THREADS, 0, stream>>>(curb, nxtb, Wb, b, x, f);
        check_kernel<<<1, 1, 0, stream>>>(f, i);
    }
    extract_kernel<<<(BB * NUM_OUT + 255) / 256, 256, 0, stream>>>(buf0, buf1, f, out);
}
